// MS2GNN_77386720739717
// MI455X (gfx1250) — matrix operand order repaired and checked by reading, not compiled, not run on hardware
//
#include <hip/hip_runtime.h>
#include <hip/hip_bf16.h>

// ---------------------------------------------------------------------------
// Types for WMMA fragments (gfx1250, wave32)
// ---------------------------------------------------------------------------
typedef __attribute__((ext_vector_type(16))) _Float16 v16h;
typedef __attribute__((ext_vector_type(8)))  float    v8f;
typedef __attribute__((ext_vector_type(2)))  _Float16 h2v;
typedef __attribute__((ext_vector_type(4)))  unsigned u4v;

union F16Frag {
  u4v      q[2];   // two 16-byte loads
  v16h     v;      // WMMA operand
  h2v      h2[8];  // packed f16 math
  unsigned u[8];
};

__device__ __forceinline__ float sigmoidf_(float x) {
  return 1.0f / (1.0f + __expf(-x));
}

// ---------------------------------------------------------------------------
// LSTM: one block (128 threads) per sample; loops T steps, h/c recurrence
// kept in LDS/registers. Writes f32 and f16 flattened outputs [N, T*H].
// ---------------------------------------------------------------------------
__global__ __launch_bounds__(128) void lstm_kernel(
    const float* __restrict__ x,    // [N, T, F]
    const float* __restrict__ Wih,  // [4H, F]
    const float* __restrict__ Whh,  // [4H, H]
    const float* __restrict__ b,    // [4H]
    float* __restrict__ Z,          // [N, T*H]
    _Float16* __restrict__ Zh,      // [N, T*H]
    int T, int F)
{
  __shared__ float sh_h[128];
  __shared__ float sh_x[64];
  const int n = blockIdx.x, tid = threadIdx.x;
  float c = 0.0f;
  sh_h[tid] = 0.0f;
  const float* xr = x + (size_t)n * T * F;
  const float bi = b[tid], bf = b[128 + tid], bg = b[256 + tid], bo = b[384 + tid];
  const float* wi0 = Wih + (size_t)(tid)       * F;
  const float* wi1 = Wih + (size_t)(128 + tid) * F;
  const float* wi2 = Wih + (size_t)(256 + tid) * F;
  const float* wi3 = Wih + (size_t)(384 + tid) * F;
  const float* wh0 = Whh + (size_t)(tid)       * 128;
  const float* wh1 = Whh + (size_t)(128 + tid) * 128;
  const float* wh2 = Whh + (size_t)(256 + tid) * 128;
  const float* wh3 = Whh + (size_t)(384 + tid) * 128;

  for (int t = 0; t < T; ++t) {
    if (tid < F) sh_x[tid] = xr[t * F + tid];
    __syncthreads();
    float zi = bi, zf = bf, zg = bg, zo = bo;
    for (int f = 0; f < F; ++f) {
      const float xv = sh_x[f];
      zi += wi0[f] * xv; zf += wi1[f] * xv; zg += wi2[f] * xv; zo += wi3[f] * xv;
    }
    for (int k = 0; k < 128; ++k) {
      const float hv = sh_h[k];
      zi += wh0[k] * hv; zf += wh1[k] * hv; zg += wh2[k] * hv; zo += wh3[k] * hv;
    }
    c = sigmoidf_(zf) * c + sigmoidf_(zi) * tanhf(zg);
    const float h = sigmoidf_(zo) * tanhf(c);
    Z [(size_t)n * T * 128 + t * 128 + tid] = h;
    Zh[(size_t)n * T * 128 + t * 128 + tid] = (_Float16)h;
    __syncthreads();            // all reads of sh_h done
    sh_h[tid] = h;
  }
}

// ---------------------------------------------------------------------------
// W1 [H=128, D] f32  ->  W1t [D, 128] f16 (transposed for B-fragment loads)
// ---------------------------------------------------------------------------
__global__ void w1t_kernel(const float* __restrict__ W1, _Float16* __restrict__ W1t, int D)
{
  const int idx = blockIdx.x * 256 + threadIdx.x;
  const int d = idx >> 7, h = idx & 127;
  W1t[idx] = (_Float16)W1[(size_t)h * D + d];
}

// ---------------------------------------------------------------------------
// adj kernel: the compute-dominant pairwise-diff MLP.
// Block = 256 thr = 8 waves; covers 8 i-rows x 16 j-cols.
// Wave w: i = blockIdx.y*8+w, computes 16(j) x 128(h) via 8 f32 accumulators,
// K-loop over D in chunks of 32 with v_wmma_f32_16x16x32_f16.
//
// B panel (W1t rows k0..k0+31, one contiguous 8KB block shared by all 8
// waves) is LDS double-buffered: iteration k issues global loads for panel
// k+1, runs 8 WMMAs against LDS panel k (16-bit ds loads, short dscnt waits),
// then ds_stores panel k+1 and barriers. LDS rows padded to 136 halfs so
// the 16-bit fragment reads spread across the 64 banks.
// A = |x_i - x_j| built on the fly with packed f16 sub + sign mask.
// ---------------------------------------------------------------------------
__global__ __launch_bounds__(256) void adj_kernel(
    const _Float16* __restrict__ xh,   // [n, D] f16
    int D, int n,
    const _Float16* __restrict__ w1t,  // [D, 128] f16
    const float* __restrict__ b1,      // [128]
    const float* __restrict__ w2,      // [128]
    const float* __restrict__ b2p,     // [1]
    float* __restrict__ adj, int pitch)
{
  __shared__ _Float16 bsh[2][32][136];   // padded pitch: 272B = 68 dwords

  const int tid  = threadIdx.x;
  const int lane = tid & 31;
  const int wv   = tid >> 5;
  const int i    = blockIdx.y * 8 + wv;      // always < 200 (25*8)
  const int j0   = blockIdx.x * 16;
  const int m    = lane & 15;
  const int hi   = lane >> 4;
  const int dlt  = hi * 8;                   // K offset for high half-wave
  int jc = j0 + m; jc = (jc < n) ? jc : (n - 1);

  const _Float16* xi = xh + (size_t)i  * D;
  const _Float16* xj = xh + (size_t)jc * D;

  // staging indices: thread t covers halfs [t*8, t*8+8) and [2048+t*8, ...)
  const int srow0 = tid >> 4;            // 0..15
  const int scol  = (tid & 15) * 8;      // 0..120

  v8f acc[8];
  #pragma unroll
  for (int ht = 0; ht < 8; ++ht) {
    const float bv = b1[ht * 16 + m];        // C(r, lane) -> h = ht*16 + m
    #pragma unroll
    for (int e = 0; e < 8; ++e) acc[ht][e] = bv;
  }

  const int nk = D >> 5;                 // number of 32-wide K chunks (even)

  // --- prologue: stage panel 0 ---
  {
    const _Float16* src = w1t;
    u4v g0 = *(const u4v*)(src + tid * 8);
    u4v g1 = *(const u4v*)(src + 2048 + tid * 8);
    *(u4v*)&bsh[0][srow0][scol]      = g0;
    *(u4v*)&bsh[0][16 + srow0][scol] = g1;
  }
  __syncthreads();

  #pragma unroll 2
  for (int kc = 0; kc < nk; ++kc) {
    const int buf  = kc & 1;
    const int k0   = kc * 32;
    const bool more = (kc + 1) < nk;

    // --- issue global loads for next panel (consumed after the WMMAs) ---
    u4v g0 = {}, g1 = {};
    if (more) {
      const _Float16* srcn = w1t + (size_t)(kc + 1) * 4096;
      g0 = *(const u4v*)(srcn + tid * 8);
      g1 = *(const u4v*)(srcn + 2048 + tid * 8);
    }
    if (kc + 2 < nk)
      __builtin_prefetch(w1t + (size_t)(kc + 2) * 4096 + tid * 8, 0, 1);

    // --- A operand: |xi - xj| in packed f16 ---
    F16Frag fi, fj, fa;
    const _Float16* pi = xi + k0 + dlt;
    const _Float16* pj = xj + k0 + dlt;
    fi.q[0] = *(const u4v*)(pi);
    fi.q[1] = *(const u4v*)(pi + 16);
    fj.q[0] = *(const u4v*)(pj);
    fj.q[1] = *(const u4v*)(pj + 16);
    #pragma unroll
    for (int e = 0; e < 8; ++e) {
      h2v d = fi.h2[e] - fj.h2[e];                       // v_pk_add_f16 (neg)
      fa.u[e] = __builtin_bit_cast(unsigned, d) & 0x7FFF7FFFu;  // |.| per f16
    }

    // --- 8 WMMAs from the LDS panel (lane%16 = N column, elements over K) ---
    #pragma unroll
    for (int ht = 0; ht < 8; ++ht) {
      F16Frag fb;
      #pragma unroll
      for (int s = 0; s < 16; ++s)
        fb.v[s] = bsh[buf][(s < 8) ? (dlt + s) : (16 + dlt + s - 8)][ht * 16 + m];
      acc[ht] = __builtin_amdgcn_wmma_f32_16x16x32_f16(
          false, fa.v, false, fb.v, (short)0, acc[ht], false, false);
    }

    // --- commit next panel to the other LDS buffer ---
    if (more) {
      *(u4v*)&bsh[buf ^ 1][srow0][scol]      = g0;
      *(u4v*)&bsh[buf ^ 1][16 + srow0][scol] = g1;
    }
    __syncthreads();
  }

  const float b2 = b2p[0];
  float wv2[8];
  #pragma unroll
  for (int ht = 0; ht < 8; ++ht) wv2[ht] = w2[ht * 16 + m];

  #pragma unroll
  for (int r = 0; r < 8; ++r) {                 // pair M = r + 8*hi
    float s = 0.0f;
    #pragma unroll
    for (int ht = 0; ht < 8; ++ht) s += wv2[ht] * fmaxf(acc[ht][r], 0.0f);
    s += __shfl_xor(s, 1, 32);
    s += __shfl_xor(s, 2, 32);
    s += __shfl_xor(s, 4, 32);
    s += __shfl_xor(s, 8, 32);
    const int col = j0 + r + 8 * hi;
    if (m == 0 && col < n)
      adj[(size_t)i * pitch + col] = sigmoidf_(s + b2);
  }
}

// ---------------------------------------------------------------------------
// dinv[i] = rsqrt(sum_j adj[i][j])
// ---------------------------------------------------------------------------
__global__ __launch_bounds__(256) void dinv_kernel(
    const float* __restrict__ adj, int pitch, int n, float* __restrict__ dinv)
{
  __shared__ float red[256];
  const int i = blockIdx.x;
  float s = 0.0f;
  for (int j = threadIdx.x; j < n; j += 256) s += adj[(size_t)i * pitch + j];
  red[threadIdx.x] = s; __syncthreads();
  for (int st = 128; st > 0; st >>= 1) {
    if (threadIdx.x < (unsigned)st) red[threadIdx.x] += red[threadIdx.x + st];
    __syncthreads();
  }
  if (threadIdx.x == 0) dinv[i] = rsqrtf(red[0]);
}

// ---------------------------------------------------------------------------
// XW[i][o] = sum_d x[i][d] * Wc[d][o]   (Wc is [D, 128] row-major)
// ---------------------------------------------------------------------------
__global__ __launch_bounds__(128) void xw_kernel(
    const float* __restrict__ x, const float* __restrict__ Wc, int D,
    float* __restrict__ xw)
{
  __shared__ float xs[128];
  const int i = blockIdx.x, o = threadIdx.x;
  float s = 0.0f;
  const float* xr = x + (size_t)i * D;
  for (int d0 = 0; d0 < D; d0 += 128) {
    __syncthreads();
    xs[o] = xr[d0 + o];
    __syncthreads();
    #pragma unroll 4
    for (int dd = 0; dd < 128; ++dd) s += xs[dd] * Wc[(size_t)(d0 + dd) * 128 + o];
  }
  xw[(size_t)i * 128 + o] = s;
}

// ---------------------------------------------------------------------------
// out[i][o] = leaky_relu( dinv[i] * sum_j adj[i][j]*dinv[j]*XW[j][o] + bc[o] )
// Also emits f16 copy (next layer's adj input).
// ---------------------------------------------------------------------------
__global__ __launch_bounds__(128) void axw_kernel(
    const float* __restrict__ adj, int pitch, int n,
    const float* __restrict__ dinv, const float* __restrict__ xw,
    const float* __restrict__ bc,
    float* __restrict__ outf, _Float16* __restrict__ outh)
{
  __shared__ float as[128];
  const int i = blockIdx.x, o = threadIdx.x;
  float s = 0.0f;
  for (int j0 = 0; j0 < n; j0 += 128) {
    int jn = n - j0; if (jn > 128) jn = 128;
    __syncthreads();
    if (o < jn) as[o] = adj[(size_t)i * pitch + j0 + o] * dinv[j0 + o];
    __syncthreads();
    for (int jj = 0; jj < jn; ++jj) s += as[jj] * xw[(size_t)(j0 + jj) * 128 + o];
  }
  float v = dinv[i] * s + bc[o];
  v = (v > 0.0f) ? v : 0.2f * v;
  outf[(size_t)i * 128 + o] = v;
  outh[(size_t)i * 128 + o] = (_Float16)v;
}

// ---------------------------------------------------------------------------
// Zsh = 0.5 * (Zsh_a + Zsh_e)
// ---------------------------------------------------------------------------
__global__ void avg_kernel(const float* __restrict__ a, const float* __restrict__ b,
                           float* __restrict__ out)
{
  const int idx = blockIdx.x * 256 + threadIdx.x;
  out[idx] = 0.5f * (a[idx] + b[idx]);
}

// ---------------------------------------------------------------------------
// Attention fusion: per-row scalar gates tanh(w . z + b), then weighted sum.
// ---------------------------------------------------------------------------
__global__ __launch_bounds__(128) void att_kernel(
    const float* __restrict__ Zsh, const float* __restrict__ Zpa,
    const float* __restrict__ Zpe,
    const float* __restrict__ Wsh, const float* __restrict__ bsh_,
    const float* __restrict__ Wa,  const float* __restrict__ ba,
    const float* __restrict__ We,  const float* __restrict__ be,
    float* __restrict__ fused)
{
  __shared__ float red[128];
  __shared__ float t[3];
  const int i = blockIdx.x, o = threadIdx.x;
  const float zsh = Zsh[(size_t)i * 128 + o];
  const float zpa = Zpa[(size_t)i * 128 + o];
  const float zpe = Zpe[(size_t)i * 128 + o];

  red[o] = Wsh[o] * zsh; __syncthreads();
  for (int st = 64; st > 0; st >>= 1) { if (o < st) red[o] += red[o + st]; __syncthreads(); }
  if (o == 0) t[0] = tanhf(red[0] + bsh_[0]); __syncthreads();

  red[o] = Wa[o] * zpa; __syncthreads();
  for (int st = 64; st > 0; st >>= 1) { if (o < st) red[o] += red[o + st]; __syncthreads(); }
  if (o == 0) t[1] = tanhf(red[0] + ba[0]); __syncthreads();

  red[o] = We[o] * zpe; __syncthreads();
  for (int st = 64; st > 0; st >>= 1) { if (o < st) red[o] += red[o + st]; __syncthreads(); }
  if (o == 0) t[2] = tanhf(red[0] + be[0]); __syncthreads();

  fused[(size_t)i * 128 + o] = t[0] * zsh + t[1] * zpa + t[2] * zpe;
}

// ---------------------------------------------------------------------------
// Classifier: hidden = relu(fused @ W1^T + b1) [64], logits = hidden @ W2^T + b2
// ---------------------------------------------------------------------------
__global__ __launch_bounds__(64) void cls_kernel(
    const float* __restrict__ fused,
    const float* __restrict__ W1, const float* __restrict__ b1,
    const float* __restrict__ W2, const float* __restrict__ b2,
    float* __restrict__ out)
{
  __shared__ float fr[128];
  __shared__ float hs[64];
  const int i = blockIdx.x, o = threadIdx.x;
  fr[o]      = fused[(size_t)i * 128 + o];
  fr[o + 64] = fused[(size_t)i * 128 + o + 64];
  __syncthreads();
  float s = b1[o];
  for (int k = 0; k < 128; ++k) s += W1[(size_t)o * 128 + k] * fr[k];
  hs[o] = fmaxf(s, 0.0f);
  __syncthreads();
  if (o < 2) {
    float l = b2[o];
    for (int j = 0; j < 64; ++j) l += W2[(size_t)o * 64 + j] * hs[j];
    out[(size_t)i * 2 + o] = l;
  }
}

// ---------------------------------------------------------------------------
// Host orchestration
// ---------------------------------------------------------------------------
extern "C" void kernel_launch(void* const* d_in, const int* in_sizes, int n_in,
                              void* d_out, int out_size, void* d_ws, size_t ws_size,
                              hipStream_t stream)
{
  (void)in_sizes; (void)n_in; (void)out_size; (void)ws_size;
  constexpr int N = 200, H = 128;
  constexpr int Ta = 20, Fa = 64, Te = 10, Fe = 32;
  constexpr int DA = Ta * H;  // 2560
  constexpr int DE = Te * H;  // 1280
  constexpr int PITCH = 208;  // 13 * 16 (padded j)

  auto F = [&](int i) { return (const float*)d_in[i]; };

  // --- workspace allocator (256-byte aligned) ---
  size_t off = 0;
  auto alloc = [&](size_t bytes) -> void* {
    size_t a = (off + 255) & ~(size_t)255;
    off = a + bytes;
    return (void*)((char*)d_ws + a);
  };

  float*    Za   = (float*)   alloc((size_t)N * DA * 4);
  _Float16* Zah  = (_Float16*)alloc((size_t)N * DA * 2);
  float*    Ze   = (float*)   alloc((size_t)N * DE * 4);
  _Float16* Zeh  = (_Float16*)alloc((size_t)N * DE * 2);

  const int w1D[8] = {DA, 128, DE, 128, DA, 128, DE, 128};
  _Float16* w1t[8];
  for (int l = 0; l < 8; ++l) w1t[l] = (_Float16*)alloc((size_t)w1D[l] * 128 * 2);

  float* adj   = (float*)alloc((size_t)N * PITCH * 4);
  float* dinv  = (float*)alloc((size_t)N * 4);
  float* xwbuf = (float*)alloc((size_t)N * 128 * 4);

  float* midf[4]; _Float16* midh[4]; float* outf[4]; _Float16* outh[4];
  for (int k = 0; k < 4; ++k) {
    midf[k] = (float*)   alloc((size_t)N * 128 * 4);
    midh[k] = (_Float16*)alloc((size_t)N * 128 * 2);
    outf[k] = (float*)   alloc((size_t)N * 128 * 4);
    outh[k] = (_Float16*)alloc((size_t)N * 128 * 2);
  }
  float* Zsh   = (float*)alloc((size_t)N * 128 * 4);
  float* fused = (float*)alloc((size_t)N * 128 * 4);

  // --- LSTMs ---
  lstm_kernel<<<N, 128, 0, stream>>>((const float*)d_in[0], F(2), F(3), F(4),
                                     Za, Zah, Ta, Fa);
  lstm_kernel<<<N, 128, 0, stream>>>((const float*)d_in[1], F(5), F(6), F(7),
                                     Ze, Zeh, Te, Fe);

  // --- one DynGCN layer ---
  auto run_layer = [&](const float* xf, const _Float16* xhp, int D, int pidx,
                       _Float16* w1tb, float* of, _Float16* oh) {
    w1t_kernel<<<(D * 128) / 256, 256, 0, stream>>>(F(pidx), w1tb, D);
    adj_kernel<<<dim3(13, 25), 256, 0, stream>>>(xhp, D, N, w1tb,
                                                 F(pidx + 1), F(pidx + 2), F(pidx + 3),
                                                 adj, PITCH);
    dinv_kernel<<<N, 256, 0, stream>>>(adj, PITCH, N, dinv);
    xw_kernel<<<N, 128, 0, stream>>>(xf, F(pidx + 4), D, xwbuf);
    axw_kernel<<<N, 128, 0, stream>>>(adj, PITCH, N, dinv, xwbuf, F(pidx + 5), of, oh);
  };

  // nets: shared_a(8), shared_e(20), spec_a(32), spec_e(44); each 2 layers (+6)
  const int    pbase[4] = {8, 20, 32, 44};
  const float* xin [4]  = {Za, Ze, Za, Ze};
  const _Float16* xinh[4] = {Zah, Zeh, Zah, Zeh};
  const int    Din [4]  = {DA, DE, DA, DE};

  for (int k = 0; k < 4; ++k) {
    run_layer(xin[k], xinh[k], Din[k], pbase[k],     w1t[2 * k],     midf[k], midh[k]);
    run_layer(midf[k], midh[k], 128,   pbase[k] + 6, w1t[2 * k + 1], outf[k], outh[k]);
  }

  // --- fusion + classifier ---
  avg_kernel<<<(N * 128) / 256, 256, 0, stream>>>(outf[0], outf[1], Zsh);
  att_kernel<<<N, 128, 0, stream>>>(Zsh, outf[2], outf[3],
                                    F(56), F(57), F(58), F(59), F(60), F(61), fused);
  cls_kernel<<<N, 64, 0, stream>>>(fused, F(62), F(63), F(64), F(65), (float*)d_out);
}
